// MOEBlock_10797547782276
// MI455X (gfx1250) — matrix operand order repaired and checked by reading, not compiled, not run on hardware
//
#include <hip/hip_runtime.h>
#include <hip/hip_bf16.h>
#include <math.h>

// ---------------- problem constants ----------------
#define BSZ   4
#define NSEQ  1024
#define DMODEL 1024
#define NHEAD 16
#define HDIM  64
#define NEXP  23
#define TOPK  3
#define HID   768
#define TTOK  (BSZ * NSEQ)    // 4096
#define LNEPS 1e-5f

// ---------------- WMMA fragment helpers (gfx1250 wave32) ----------------
typedef __attribute__((ext_vector_type(16))) _Float16 v16h;
typedef __attribute__((ext_vector_type(8)))  float    v8f;

struct HFrag { union { v16h v; _Float16 h[16]; uint32_t u[8]; }; };
struct FFrag { union { v8f v; float f[8]; }; };

__device__ inline v8f wmma_f16(v16h a, v16h b, v8f c) {
  // D = A(16x32 f16) * B(32x16 f16) + C(16x16 f32)
  return __builtin_amdgcn_wmma_f32_16x16x32_f16(false, a, false, b, (short)0, c,
                                                false, false);
}

// A fragment 16x32 f16 from row-major storage (ld in elements).
// ISA layout: lanes 0-15 row m=lane, K 0..7 (v0..3) & 16..23 (v4..7);
//             lanes 16-31 same rows, K 8..15 & 24..31. Pairs contiguous -> b128 loads.
__device__ inline HFrag load_afrag(const _Float16* s, int ld) {
  const int lane = threadIdx.x & 31;
  const int m = lane & 15;
  const int kb = (lane & 16) ? 8 : 0;
  HFrag r;
#pragma unroll
  for (int v = 0; v < 8; ++v) {
    const int k = ((v & 4) ? 16 : 0) + kb + (v & 3) * 2;
    r.u[v] = *(const uint32_t*)(s + (long long)m * ld + k);
  }
  return r;
}

// B fragment 32x16 f16 from TRANSPOSED storage: B[k][n] held at s[n*ld + k].
// Lanes 0-15: col n=lane, K 0..7 & 16..23; lanes 16-31: same cols, K 8..15 & 24..31.
// Pairs contiguous in k -> b128 loads (fast path; avoids ds_load_u16 storms).
__device__ inline HFrag load_bfrag_trans(const _Float16* s, int ld) {
  const int lane = threadIdx.x & 31;
  const int n = lane & 15;
  const int kb = (lane & 16) ? 8 : 0;
  HFrag r;
#pragma unroll
  for (int v = 0; v < 8; ++v) {
    r.u[v] = *(const uint32_t*)(s + (long long)n * ld + ((v & 4) ? 16 : 0) + kb + (v & 3) * 2);
  }
  return r;
}

__device__ inline float gelu_f(float x) {
  return 0.5f * x * (1.0f + erff(x * 0.70710678118654752f));
}

// ---------------- elementwise / small kernels ----------------
__global__ void f32_to_f16_kernel(const float* __restrict__ src,
                                  _Float16* __restrict__ dst, long long n) {
  long long i = (long long)blockIdx.x * blockDim.x + threadIdx.x;
  const long long stride = (long long)gridDim.x * blockDim.x;
  for (; i < n; i += stride) dst[i] = (_Float16)src[i];
}

__global__ __launch_bounds__(256)
void ln_kernel(const float* __restrict__ x, const float* __restrict__ g,
               const float* __restrict__ b, _Float16* __restrict__ out) {
  const int t = blockIdx.x;
  const int tid = threadIdx.x;
  __shared__ float red[256];
  const float* row = x + (long long)t * DMODEL;
  float vals[4];
  float s = 0.f, sq = 0.f;
#pragma unroll
  for (int i = 0; i < 4; ++i) {
    vals[i] = row[tid + i * 256];
    s += vals[i];
    sq += vals[i] * vals[i];
  }
  red[tid] = s; __syncthreads();
  for (int o = 128; o > 0; o >>= 1) { if (tid < o) red[tid] += red[tid + o]; __syncthreads(); }
  const float mean = red[0] * (1.0f / DMODEL);
  __syncthreads();
  red[tid] = sq; __syncthreads();
  for (int o = 128; o > 0; o >>= 1) { if (tid < o) red[tid] += red[tid + o]; __syncthreads(); }
  const float var = red[0] * (1.0f / DMODEL) - mean * mean;
  const float inv = rsqrtf(var + LNEPS);
#pragma unroll
  for (int i = 0; i < 4; ++i) {
    const int d = tid + i * 256;
    out[(long long)t * DMODEL + d] = (_Float16)((vals[i] - mean) * inv * g[d] + b[d]);
  }
}

// ---------------- generic WMMA GEMM (plain / indirect / scatter) ----------------
#define GBM 128
#define GBN 128
#define GBK 64
#define G_LDA 72   // GBK + 8 pad
#define G_LDB 72   // Bs stored transposed: [n][k], row stride GBK + 8

// aMode: 0 = A row r; 1 = A row perm[off+r]/TOPK; 2 = A row off+r
// outMode: 0 = f16 out row r; 1 = f16 out row off+r; 2 = f16 out row perm[off+r];
//          3 = f32 out row r;  4 = f32 out row r + residual
// act: 0 none, 1 exact GELU
__global__ __launch_bounds__(256)
void gemm_kernel(const _Float16* __restrict__ A, int lda,
                 const _Float16* __restrict__ Bw, long long bStride, int ldb,
                 const float* __restrict__ bias, int biasStride,
                 int M, int Ndim, int Kdim,
                 int aMode, int outMode, int act,
                 const int* __restrict__ perm, const int* __restrict__ offsets,
                 _Float16* __restrict__ outH, int ldoH,
                 float* __restrict__ outF, int ldoF,
                 const float* __restrict__ resid) {
  const int e = blockIdx.z;
  int off = 0, cnt = M;
  if (aMode != 0 || outMode == 1 || outMode == 2) {
    off = offsets[e];
    cnt = offsets[e + 1] - off;
  }
  const int tm = blockIdx.x, tn = blockIdx.y;
  if (tm * GBM >= cnt) return;

  const _Float16* Bexp = Bw + bStride * (long long)e;
  const float* biasE = bias ? (bias + (long long)biasStride * e) : nullptr;

  __shared__ __align__(16) _Float16 As[GBM][G_LDA];       // row-major A tile
  __shared__ __align__(16) _Float16 Bs[GBN][G_LDB];       // TRANSPOSED B tile: [n][k]
  __shared__ int rowSrc[GBM];
  __shared__ int rowDst[GBM];

  const int tid = threadIdx.x;
  const int wave = tid >> 5;
  const int lane = tid & 31;

  if (tid < GBM) {
    const int gr = tm * GBM + tid;
    int sa = -1, so = -1;
    if (gr < cnt) {
      if (aMode == 0) sa = gr;
      else if (aMode == 1) sa = perm[off + gr] / TOPK;
      else sa = off + gr;
      if (outMode == 1) so = off + gr;
      else if (outMode == 2) so = perm[off + gr];
      else so = gr;
    }
    rowSrc[tid] = sa;
    rowDst[tid] = so;
  }
  __syncthreads();

  const int wm = wave >> 2;   // 0..1
  const int wn = wave & 3;    // 0..3
  FFrag acc[4][2];
#pragma unroll
  for (int i = 0; i < 4; ++i)
#pragma unroll
    for (int j = 0; j < 2; ++j)
#pragma unroll
      for (int v = 0; v < 8; ++v) acc[i][j].f[v] = 0.f;

  for (int k0 = 0; k0 < Kdim; k0 += GBK) {
    // stage A tile (128x64 halves = 4096 dwords), coalesced b32 loads/stores
    for (int u = tid; u < GBM * (GBK / 2); u += 256) {
      const int r = u >> 5;
      const int c2 = (u & 31) * 2;
      const int sa = rowSrc[r];
      uint32_t val = 0u;
      if (sa >= 0) val = *(const uint32_t*)(A + (long long)sa * lda + k0 + c2);
      *(uint32_t*)&As[r][c2] = val;
    }
    // stage B tile transposed (64x128 halves): coalesced global b32 reads of two
    // consecutive n, scattered as two b16 LDS stores at [n][k], [n+1][k]
    for (int u = tid; u < GBK * (GBN / 2); u += 256) {
      const int r = u >> 6;              // k within tile
      const int c2 = (u & 63) * 2;       // n within tile
      union { uint32_t u32; _Float16 h[2]; } cv;
      cv.u32 = *(const uint32_t*)(Bexp + (long long)(k0 + r) * ldb + tn * GBN + c2);
      Bs[c2][r] = cv.h[0];
      Bs[c2 + 1][r] = cv.h[1];
    }
    __syncthreads();

#pragma unroll
    for (int kk = 0; kk < GBK; kk += 32) {
      HFrag af[4];
#pragma unroll
      for (int i = 0; i < 4; ++i)
        af[i] = load_afrag(&As[wm * 64 + i * 16][kk], G_LDA);
      HFrag bf[2];
#pragma unroll
      for (int j = 0; j < 2; ++j)
        bf[j] = load_bfrag_trans(&Bs[wn * 32 + j * 16][kk], G_LDB);
#pragma unroll
      for (int i = 0; i < 4; ++i)
#pragma unroll
        for (int j = 0; j < 2; ++j)
          acc[i][j].v = wmma_f16(af[i].v, bf[j].v, acc[i][j].v);
    }
    __syncthreads();
  }

  // epilogue
  const int r8 = (lane & 16) ? 8 : 0;
  const int colW = tn * GBN + wn * 32 + (lane & 15);
#pragma unroll
  for (int i = 0; i < 4; ++i) {
#pragma unroll
    for (int j = 0; j < 2; ++j) {
#pragma unroll
      for (int v = 0; v < 8; ++v) {
        const int lrow = wm * 64 + i * 16 + v + r8;
        const int dst = rowDst[lrow];
        if (dst < 0) continue;
        const int col = colW + j * 16;
        float xv = acc[i][j].f[v];
        if (biasE) xv += biasE[col];
        if (act == 1) xv = gelu_f(xv);
        if (outMode == 3) {
          outF[(long long)dst * ldoF + col] = xv;
        } else if (outMode == 4) {
          outF[(long long)dst * ldoF + col] = xv + resid[(long long)dst * ldoF + col];
        } else {
          outH[(long long)dst * ldoH + col] = (_Float16)xv;
        }
      }
    }
  }
}

// ---------------- flash attention (WMMA, online softmax) ----------------
#define ATT_BK 32
__global__ __launch_bounds__(128)
void attention_kernel(const _Float16* __restrict__ qkv, _Float16* __restrict__ o16) {
  const int bh = blockIdx.x;
  const int b = bh / NHEAD;
  const int h = bh % NHEAD;
  const int q0 = blockIdx.y * 64;
  const int tid = threadIdx.x;
  const int wave = tid >> 5;
  const int lane = tid & 31;

  __shared__ __align__(16) _Float16 Kt[ATT_BK][HDIM + 8];   // [key][d] (row-major)
  __shared__ __align__(16) _Float16 VtT[HDIM][ATT_BK + 8];  // [d][key] (transposed)
  __shared__ float Sbuf[4][16][36];
  __shared__ __align__(16) _Float16 Pbuf[4][16][40];
  __shared__ float rowScale[4][16];
  __shared__ float rowSum[4][16];

  const int ldq = 3 * DMODEL;
  const _Float16* qbase =
      qkv + (long long)(b * NSEQ + q0 + wave * 16) * ldq + h * HDIM;
  const HFrag aQ0 = load_afrag(qbase, ldq);        // d = 0..31
  const HFrag aQ1 = load_afrag(qbase + 32, ldq);   // d = 32..63

  FFrag accO[4];
#pragma unroll
  for (int f = 0; f < 4; ++f)
#pragma unroll
    for (int v = 0; v < 8; ++v) accO[f].f[v] = 0.f;
  float m_row = -INFINITY;  // valid in lanes 0..15 (row = lane)
  float l_row = 0.f;

  for (int kt = 0; kt < NSEQ; kt += ATT_BK) {
    // stage K (row-major) and V (transposed) tiles
    for (int u = tid; u < ATT_BK * (HDIM / 2); u += 128) {
      const int r = u >> 5;              // key within tile
      const int c2 = (u & 31) * 2;       // d
      const _Float16* kp =
          qkv + (long long)(b * NSEQ + kt + r) * ldq + DMODEL + h * HDIM + c2;
      *(uint32_t*)&Kt[r][c2] = *(const uint32_t*)kp;
      union { uint32_t u32; _Float16 h[2]; } cv;
      cv.u32 = *(const uint32_t*)(kp + DMODEL);
      VtT[c2][r] = cv.h[0];
      VtT[c2 + 1][r] = cv.h[1];
    }
    __syncthreads();

    // S(16x32) = Q(16x64) * K_tile^T, staged to Sbuf
#pragma unroll
    for (int j = 0; j < 2; ++j) {
      FFrag c;
#pragma unroll
      for (int v = 0; v < 8; ++v) c.f[v] = 0.f;
      HFrag b0 = load_bfrag_trans(&Kt[j * 16][0], HDIM + 8);
      c.v = wmma_f16(aQ0.v, b0.v, c.v);
      HFrag b1 = load_bfrag_trans(&Kt[j * 16][32], HDIM + 8);
      c.v = wmma_f16(aQ1.v, b1.v, c.v);
      const int r8 = (lane & 16) ? 8 : 0;
#pragma unroll
      for (int v = 0; v < 8; ++v)
        Sbuf[wave][v + r8][(lane & 15) + j * 16] = c.f[v] * 0.125f;  // 1/sqrt(64)
    }

    // online softmax; lane l (<16) owns query row l of this wave
    if (lane < 16) {
      float mt = -INFINITY;
#pragma unroll
      for (int c = 0; c < 32; ++c) mt = fmaxf(mt, Sbuf[wave][lane][c]);
      const float mnew = fmaxf(m_row, mt);
      const float ef = expf(m_row - mnew);
      float ps = 0.f;
#pragma unroll
      for (int c = 0; c < 32; ++c) {
        const float p = expf(Sbuf[wave][lane][c] - mnew);
        ps += p;
        Pbuf[wave][lane][c] = (_Float16)p;
      }
      l_row = l_row * ef + ps;
      m_row = mnew;
      rowScale[wave][lane] = ef;
    }
    // rescale accumulators (same-wave LDS ops are in order)
    {
      const int r8 = (lane & 16) ? 8 : 0;
#pragma unroll
      for (int f = 0; f < 4; ++f)
#pragma unroll
        for (int v = 0; v < 8; ++v) accO[f].f[v] *= rowScale[wave][v + r8];
    }
    // O += P(16x32) * V(32x64); V staged transposed -> contiguous B loads
    const HFrag aP = load_afrag(&Pbuf[wave][0][0], 40);
#pragma unroll
    for (int f = 0; f < 4; ++f) {
      HFrag bv = load_bfrag_trans(&VtT[f * 16][0], ATT_BK + 8);
      accO[f].v = wmma_f16(aP.v, bv.v, accO[f].v);
    }
    __syncthreads();
  }

  if (lane < 16) rowSum[wave][lane] = l_row;
  const int r8 = (lane & 16) ? 8 : 0;
#pragma unroll
  for (int f = 0; f < 4; ++f) {
#pragma unroll
    for (int v = 0; v < 8; ++v) {
      const int qrow = q0 + wave * 16 + v + r8;
      const float val = accO[f].f[v] / rowSum[wave][v + r8];
      const int col = (lane & 15) + f * 16;
      o16[(long long)(b * NSEQ + qrow) * DMODEL + h * HDIM + col] = (_Float16)val;
    }
  }
}

// ---------------- MoE routing ----------------
__global__ __launch_bounds__(256)
void gate_topk_kernel(const _Float16* __restrict__ h2, const float* __restrict__ gw,
                      const float* __restrict__ gb, const float* __restrict__ biasbuf,
                      float* __restrict__ gate, int* __restrict__ topi,
                      float* __restrict__ combw) {
  const int t = blockIdx.x;
  const int tid = threadIdx.x;
  __shared__ float xs[DMODEL];
  __shared__ float gsh[NEXP];
  for (int d = tid; d < DMODEL; d += 256) xs[d] = (float)h2[(long long)t * DMODEL + d];
  __syncthreads();
  if (tid < NEXP) {
    float acc = gb[tid];
    for (int d = 0; d < DMODEL; ++d) acc += xs[d] * gw[d * NEXP + tid];
    const float g = 1.0f / (1.0f + expf(-acc));
    gsh[tid] = g;
    gate[t * NEXP + tid] = g;
  }
  __syncthreads();
  if (tid == 0) {
    bool used[NEXP];
    for (int e = 0; e < NEXP; ++e) used[e] = false;
    int sel[TOPK];
    float tw[TOPK];
    float tsum = 0.f;
    for (int k = 0; k < TOPK; ++k) {
      float best = -1e30f;
      int bi = 0;
      for (int e = 0; e < NEXP; ++e) {
        if (used[e]) continue;
        const float sc = gsh[e] + biasbuf[e];
        if (sc > best) { best = sc; bi = e; }
      }
      used[bi] = true;
      sel[k] = bi;
      tw[k] = gsh[bi];
      tsum += tw[k];
    }
    for (int k = 0; k < TOPK; ++k) {
      topi[t * TOPK + k] = sel[k];
      combw[t * TOPK + k] = tw[k] / tsum;
    }
  }
}

__global__ void zero_counts_kernel(int* counts) {
  if (threadIdx.x < NEXP) counts[threadIdx.x] = 0;
}

__global__ void count_kernel(const int* __restrict__ topi, int* __restrict__ counts) {
  const int i = blockIdx.x * 256 + threadIdx.x;
  if (i < TTOK * TOPK) atomicAdd(&counts[topi[i]], 1);
}

__global__ void scan_kernel(const int* __restrict__ counts, int* __restrict__ offsets,
                            int* __restrict__ cursor) {
  if (threadIdx.x == 0) {
    int s = 0;
    for (int e = 0; e < NEXP; ++e) { offsets[e] = s; cursor[e] = s; s += counts[e]; }
    offsets[NEXP] = s;
  }
}

__global__ void fill_perm_kernel(const int* __restrict__ topi, int* __restrict__ cursor,
                                 int* __restrict__ perm) {
  const int i = blockIdx.x * 256 + threadIdx.x;
  if (i < TTOK * TOPK) {
    const int e = topi[i];
    const int pos = atomicAdd(&cursor[e], 1);
    perm[pos] = i;  // i = t*TOPK + slot; row position in list is irrelevant to output
  }
}

// ---------------- combine + aux ----------------
__global__ __launch_bounds__(256)
void combine_kernel(const float* __restrict__ x1, const float* __restrict__ share,
                    const _Float16* __restrict__ eout, const float* __restrict__ combw,
                    float* __restrict__ out) {
  const int t = blockIdx.x;
  const int tid = threadIdx.x;
  const float w0 = combw[t * TOPK + 0];
  const float w1 = combw[t * TOPK + 1];
  const float w2 = combw[t * TOPK + 2];
  const long long eb = (long long)t * TOPK * DMODEL;
  for (int d = tid; d < DMODEL; d += 256) {
    float v = x1[(long long)t * DMODEL + d] + share[(long long)t * DMODEL + d];
    v += w0 * (float)eout[eb + 0 * DMODEL + d];
    v += w1 * (float)eout[eb + 1 * DMODEL + d];
    v += w2 * (float)eout[eb + 2 * DMODEL + d];
    out[(long long)t * DMODEL + d] = v;
  }
}

__global__ __launch_bounds__(256)
void aux_partial_kernel(const float* __restrict__ gate, const int* __restrict__ counts,
                        float* __restrict__ auxPart) {
  const int e = blockIdx.x;
  const int tid = threadIdx.x;
  __shared__ float red[256];
  float acc = 0.f;
  for (int t = tid; t < TTOK; t += 256) {
    float s = 0.f;
    for (int j = 0; j < NEXP; ++j) s += gate[t * NEXP + j];
    acc += gate[t * NEXP + e] / s;
  }
  red[tid] = acc; __syncthreads();
  for (int o = 128; o > 0; o >>= 1) { if (tid < o) red[tid] += red[tid + o]; __syncthreads(); }
  if (tid == 0) {
    const float P = red[0] / (float)TTOK;
    const float F = (float)NEXP * (float)counts[e] / (float)(TOPK * TTOK);
    auxPart[e] = P * F;
  }
}

__global__ void aux_final_kernel(const float* __restrict__ auxPart, float* __restrict__ out) {
  if (threadIdx.x == 0) {
    float s = 0.f;
    for (int e = 0; e < NEXP; ++e) s += auxPart[e];
    out[(long long)TTOK * DMODEL] = s;
  }
}

// ---------------- host launcher ----------------
extern "C" void kernel_launch(void* const* d_in, const int* in_sizes, int n_in,
                              void* d_out, int out_size, void* d_ws, size_t ws_size,
                              hipStream_t stream) {
  const float* x        = (const float*)d_in[0];
  const float* norm1_g  = (const float*)d_in[1];
  const float* norm1_b  = (const float*)d_in[2];
  const float* qkv_w    = (const float*)d_in[3];
  const float* proj_w   = (const float*)d_in[4];
  const float* proj_b   = (const float*)d_in[5];
  const float* norm2_g  = (const float*)d_in[6];
  const float* norm2_b  = (const float*)d_in[7];
  const float* gate_w   = (const float*)d_in[8];
  const float* gate_b   = (const float*)d_in[9];
  const float* bias_buf = (const float*)d_in[10];
  const float* ew1      = (const float*)d_in[11];
  const float* eb1      = (const float*)d_in[12];
  const float* ew2      = (const float*)d_in[13];
  const float* eb2      = (const float*)d_in[14];
  const float* sw1      = (const float*)d_in[15];
  const float* sb1      = (const float*)d_in[16];
  const float* sw2      = (const float*)d_in[17];
  const float* sb2      = (const float*)d_in[18];
  float* out = (float*)d_out;

  char* w = (char*)d_ws;
  size_t off = 0;
  auto alloc = [&](size_t bytes) -> void* {
    off = (off + 255) & ~(size_t)255;
    void* p = w + off;
    off += bytes;
    return p;
  };

  _Float16* qkvW16 = (_Float16*)alloc((size_t)DMODEL * 3 * DMODEL * 2);
  _Float16* projW16 = (_Float16*)alloc((size_t)DMODEL * DMODEL * 2);
  _Float16* sw1_16 = (_Float16*)alloc((size_t)DMODEL * HID * 2);
  _Float16* sw2_16 = (_Float16*)alloc((size_t)HID * DMODEL * 2);
  _Float16* ew1_16 = (_Float16*)alloc((size_t)NEXP * DMODEL * HID * 2);
  _Float16* ew2_16 = (_Float16*)alloc((size_t)NEXP * HID * DMODEL * 2);
  _Float16* hA     = (_Float16*)alloc((size_t)TTOK * DMODEL * 2);
  _Float16* qkv16  = (_Float16*)alloc((size_t)TTOK * 3 * DMODEL * 2);
  _Float16* o16    = (_Float16*)alloc((size_t)TTOK * DMODEL * 2);
  float*    x1     = (float*)alloc((size_t)TTOK * DMODEL * 4);
  _Float16* h2     = (_Float16*)alloc((size_t)TTOK * DMODEL * 2);
  float*    gateB  = (float*)alloc((size_t)TTOK * NEXP * 4);
  float*    combw  = (float*)alloc((size_t)TTOK * TOPK * 4);
  int*      topi   = (int*)alloc((size_t)TTOK * TOPK * 4);
  int*      counts = (int*)alloc((size_t)NEXP * 4);
  int*      offsets = (int*)alloc((size_t)(NEXP + 1) * 4);
  int*      cursor = (int*)alloc((size_t)NEXP * 4);
  int*      perm   = (int*)alloc((size_t)TTOK * TOPK * 4);
  _Float16* ehid   = (_Float16*)alloc((size_t)TTOK * TOPK * HID * 2);
  _Float16* eout   = (_Float16*)alloc((size_t)TTOK * TOPK * DMODEL * 2);
  _Float16* shid   = (_Float16*)alloc((size_t)TTOK * HID * 2);
  float*    share  = (float*)alloc((size_t)TTOK * DMODEL * 4);
  float*    auxPart = (float*)alloc((size_t)NEXP * 4);
  (void)ws_size; (void)n_in; (void)in_sizes; (void)out_size;

  // 1) weight conversions fp32 -> f16
  f32_to_f16_kernel<<<2048, 256, 0, stream>>>(qkv_w, qkvW16, (long long)DMODEL * 3 * DMODEL);
  f32_to_f16_kernel<<<2048, 256, 0, stream>>>(proj_w, projW16, (long long)DMODEL * DMODEL);
  f32_to_f16_kernel<<<2048, 256, 0, stream>>>(sw1, sw1_16, (long long)DMODEL * HID);
  f32_to_f16_kernel<<<2048, 256, 0, stream>>>(sw2, sw2_16, (long long)HID * DMODEL);
  f32_to_f16_kernel<<<4096, 256, 0, stream>>>(ew1, ew1_16, (long long)NEXP * DMODEL * HID);
  f32_to_f16_kernel<<<4096, 256, 0, stream>>>(ew2, ew2_16, (long long)NEXP * HID * DMODEL);

  // 2) LN1
  ln_kernel<<<TTOK, 256, 0, stream>>>(x, norm1_g, norm1_b, hA);

  // 3) QKV GEMM: [T,D] x [D,3D] -> f16 [T,3D]
  gemm_kernel<<<dim3(TTOK / GBM, 3 * DMODEL / GBN, 1), 256, 0, stream>>>(
      hA, DMODEL, qkvW16, 0, 3 * DMODEL, nullptr, 0,
      TTOK, 3 * DMODEL, DMODEL, 0, 0, 0, nullptr, nullptr,
      qkv16, 3 * DMODEL, nullptr, 0, nullptr);

  // 4) flash attention
  attention_kernel<<<dim3(BSZ * NHEAD, NSEQ / 64), 128, 0, stream>>>(qkv16, o16);

  // 5) proj GEMM + bias + residual -> x1 (fp32)
  gemm_kernel<<<dim3(TTOK / GBM, DMODEL / GBN, 1), 256, 0, stream>>>(
      o16, DMODEL, projW16, 0, DMODEL, proj_b, 0,
      TTOK, DMODEL, DMODEL, 0, 4, 0, nullptr, nullptr,
      nullptr, 0, x1, DMODEL, x);

  // 6) LN2
  ln_kernel<<<TTOK, 256, 0, stream>>>(x1, norm2_g, norm2_b, h2);

  // 7) gate + sigmoid + top-3
  gate_topk_kernel<<<TTOK, 256, 0, stream>>>(h2, gate_w, gate_b, bias_buf,
                                             gateB, topi, combw);

  // 8) routing tables (integer atomics only; fp results remain deterministic)
  zero_counts_kernel<<<1, 32, 0, stream>>>(counts);
  count_kernel<<<(TTOK * TOPK + 255) / 256, 256, 0, stream>>>(topi, counts);
  scan_kernel<<<1, 32, 0, stream>>>(counts, offsets, cursor);
  fill_perm_kernel<<<(TTOK * TOPK + 255) / 256, 256, 0, stream>>>(topi, cursor, perm);

  // 9) expert FFN1: gathered rows of h2 -> gelu -> ehid (contiguous per expert)
  gemm_kernel<<<dim3(TTOK / GBM, HID / GBN, NEXP), 256, 0, stream>>>(
      h2, DMODEL, ew1_16, (long long)DMODEL * HID, HID, eb1, HID,
      TTOK, HID, DMODEL, 1, 1, 1, perm, offsets,
      ehid, HID, nullptr, 0, nullptr);

  // 10) expert FFN2: ehid -> scatter to eout[(t*K+slot)]
  gemm_kernel<<<dim3(TTOK / GBM, DMODEL / GBN, NEXP), 256, 0, stream>>>(
      ehid, HID, ew2_16, (long long)HID * DMODEL, DMODEL, eb2, DMODEL,
      TTOK, DMODEL, HID, 2, 2, 0, perm, offsets,
      eout, DMODEL, nullptr, 0, nullptr);

  // 11) shared expert MLP
  gemm_kernel<<<dim3(TTOK / GBM, HID / GBN, 1), 256, 0, stream>>>(
      h2, DMODEL, sw1_16, 0, HID, sb1, 0,
      TTOK, HID, DMODEL, 0, 0, 1, nullptr, nullptr,
      shid, HID, nullptr, 0, nullptr);
  gemm_kernel<<<dim3(TTOK / GBM, DMODEL / GBN, 1), 256, 0, stream>>>(
      shid, HID, sw2_16, 0, DMODEL, sb2, 0,
      TTOK, DMODEL, HID, 0, 3, 0, nullptr, nullptr,
      nullptr, 0, share, DMODEL, nullptr);

  // 12) combine: out = x1 + share + sum_k combw * eout
  combine_kernel<<<TTOK, 256, 0, stream>>>(x1, share, eout, combw, out);

  // 13) aux loss (fixed-order reductions)
  aux_partial_kernel<<<NEXP, 256, 0, stream>>>(gateB, counts, auxPart);
  aux_final_kernel<<<1, 32, 0, stream>>>(auxPart, out);
}
